// Model_67645734912047
// MI455X (gfx1250) — matrix operand order repaired and checked by reading, not compiled, not run on hardware
//
#include <hip/hip_runtime.h>
#include <hip/hip_bf16.h>
#include <math.h>

// ---------------------------------------------------------------------------
// coRNN-ODE + rDFT magnitude + MLP readout for MI455X (gfx1250).
// f32 convs/ODE, f16 streamed tensors, v_wmma_f32_16x16x32_f16 with f32 acc.
// B matrices pre-packed to fragment-native layout (2 x b128 per lane).
// MLP GEMM stages packed B via global_load_async_to_lds_b128 (ASYNCcnt).
// ---------------------------------------------------------------------------

typedef __attribute__((ext_vector_type(16))) _Float16 v16h;
typedef __attribute__((ext_vector_type(8)))  float    v8f;

#define BB   16
#define SS   64
#define HW   4096
#define CC   16
#define TT   50
#define KK   26             // rfft bins
#define FEAT 416            // CC*KK = 13*32
#define HID  256
#define NCLS 7
#define MROWS (BB*HW*CC)    // 1048576 rows of the DFT GEMM
#define MPIX  (BB*HW)       // 65536 rows of the MLP

// Packed-B fragment load: lane's 16 halves are 32 contiguous bytes.
__device__ __forceinline__ v16h load_b_packed(const _Float16* __restrict__ P,
                                              int tileIdx, int lane) {
  const _Float16* p = P + ((size_t)tileIdx * 32 + lane) * 16;
  union { uint4 u[2]; v16h h; } u;
  u.u[0] = *(const uint4*)p;
  u.u[1] = *(const uint4*)(p + 8);
  return u.h;
}

// A fragment (row-major, lda in halves): two contiguous 16B chunks per lane.
__device__ __forceinline__ v16h load_a_b128(const _Float16* __restrict__ arow,
                                            int k0, int hi8) {
  union { uint4 u[2]; v16h h; } u;
  const _Float16* p = arow + k0 + hi8;
  u.u[0] = *(const uint4*)p;
  u.u[1] = *(const uint4*)(p + 16);
  return u.h;
}

// ------------------------------ kernels ------------------------------------

// 3x3 'SAME' zero-pad conv, NCHW/OIHW. mode 0=relu 1=tanh.
__global__ void conv3x3(const float* __restrict__ in, const float* __restrict__ w,
                        const float* __restrict__ bias, float* __restrict__ out,
                        int Ci, int mode) {
  __shared__ float ws[CC * 9];
  int co = blockIdx.y, b = blockIdx.z;
  int tid = threadIdx.x;
  for (int i = tid; i < Ci * 9; i += 256) ws[i] = w[co * Ci * 9 + i];
  __syncthreads();
  int row = blockIdx.x * 4 + (tid >> 6);
  int col = tid & 63;
  float acc = bias[co];
  for (int ci = 0; ci < Ci; ++ci) {
    const float* ip = in + ((size_t)b * Ci + ci) * HW;
#pragma unroll
    for (int dy = -1; dy <= 1; ++dy) {
      int r = row + dy;
      if ((unsigned)r >= (unsigned)SS) continue;
#pragma unroll
      for (int dx = -1; dx <= 1; ++dx) {
        int c2 = col + dx;
        if ((unsigned)c2 >= (unsigned)SS) continue;
        acc += ws[ci * 9 + (dy + 1) * 3 + (dx + 1)] * ip[r * 64 + c2];
      }
    }
  }
  acc = mode ? tanhf(acc) : fmaxf(acc, 0.f);
  out[((size_t)b * CC + co) * HW + row * 64 + col] = acc;
}

// coRNN ODE scan: one block per (b,c) plane (wy is block-diagonal ->
// depthwise circular 9-tap stencil). hy LDS-resident; y written TIME-MAJOR
// ybuf[t*MROWS + blk*HW + p] so the f16 stores are fully coalesced.
__global__ void ode_scan(const float* __restrict__ hy0, const float* __restrict__ om,
                         const float* __restrict__ al, const float* __restrict__ wy,
                         _Float16* __restrict__ ybuf) {
  __shared__ float hs[HW];
  int blk = blockIdx.x;             // b*16 + c
  int c = blk & 15;
  int tid = threadIdx.x;
  float tap[9];
#pragma unroll
  for (int j = 0; j < 9; ++j) tap[j] = wy[(c * CC + c) * 9 + j];
  float hz[16], omg[16], alf[16];
  size_t base = (size_t)blk * HW;
#pragma unroll
  for (int i = 0; i < 16; ++i) {
    int p = i * 256 + tid;
    hs[p]  = hy0[base + p];
    omg[i] = om[base + p];
    alf[i] = al[base + p];
    hz[i]  = 0.f;
  }
  __syncthreads();
  for (int t = 0; t < TT; ++t) {
    float nhy[16];
#pragma unroll
    for (int i = 0; i < 16; ++i) {
      int p = i * 256 + tid;
      int r = p >> 6, cl = p & 63;
      int rm = (r + 63) & 63, rp = (r + 1) & 63;
      int cm = (cl + 63) & 63, cp = (cl + 1) & 63;
      float s =
          tap[0] * hs[rm * 64 + cm] + tap[1] * hs[rm * 64 + cl] + tap[2] * hs[rm * 64 + cp] +
          tap[3] * hs[r  * 64 + cm] + tap[4] * hs[r  * 64 + cl] + tap[5] * hs[r  * 64 + cp] +
          tap[6] * hs[rp * 64 + cm] + tap[7] * hs[rp * 64 + cl] + tap[8] * hs[rp * 64 + cp];
      float sf  = tanhf(s);
      float hyc = hs[r * 64 + cl];
      float nz  = hz[i] + 0.5f * (sf - omg[i] * hyc - alf[i] * hz[i]);
      hz[i]  = nz;
      nhy[i] = hyc + 0.5f * nz;
    }
    __syncthreads();
#pragma unroll
    for (int i = 0; i < 16; ++i) {
      int p = i * 256 + tid;
      hs[p] = nhy[i];
      ybuf[(size_t)t * MROWS + base + p] = (_Float16)nhy[i];
    }
    __syncthreads();
  }
}

// DFT matrix generated DIRECTLY in packed-B fragment layout:
// P[((kt*4+nt)*32+lane)*16+i] = Bd[k = kt*32+(lane>>4)*8+i+(i>=8?8:0)][n = nt*16+(lane&15)]
// cols 0..25 = cos, 26..51 = sin, rows t>=50 / cols>=52 zero (nulls K-pad).
__global__ void dft_init(_Float16* __restrict__ P) {
  int tid = threadIdx.x;                 // 256 = 8 tiles * 32 lanes
  int lane = tid & 31, tile = tid >> 5;  // tile = kt*4 + nt
  int kt = tile >> 2, nt = tile & 3;
  const float w0 = 6.28318530717958647692f / (float)TT;
  int n = nt * 16 + (lane & 15);
#pragma unroll
  for (int i = 0; i < 16; ++i) {
    int k = kt * 32 + (lane >> 4) * 8 + i + (i >= 8 ? 8 : 0);   // time index
    float v = 0.f;
    if (k < TT) {
      if (n < KK)          v = cosf(w0 * (float)n * (float)k);
      else if (n < 2 * KK) v = sinf(w0 * (float)(n - KK) * (float)k);
    }
    P[(size_t)tid * 16 + i] = (_Float16)v;
  }
}

// rDFT magnitude: 256-thread blocks, 128 rows each. Coalesced b32 loads of
// time-major y, transpose into LDS (pitch 72 halves = 144B, 16B-aligned),
// A fragments via 2x ds_load_b128, B via packed global b128, WMMA, then
// |X_k| into feature matrix F (MPIX x 416 f16).
#define YT_PITCH 72
__global__ void dft_mag(const _Float16* __restrict__ Y, const _Float16* __restrict__ BdP,
                        _Float16* __restrict__ F) {
  __shared__ __align__(16) _Float16 yt[128 * YT_PITCH];   // 18 KB
  __shared__ __align__(16) float dstage[8 * 16 * 64];     // 32 KB
  int tid = threadIdx.x, lane = tid & 31, wv = tid >> 5;
  int rowB = blockIdx.x * 128;
#pragma unroll
  for (int it = 0; it < 16; ++it) {
    int idx = it * 256 + tid;       // 64 k-slices x 64 row-pairs
    int kidx = idx >> 6;
    int rp = (idx & 63) * 2;
    union { unsigned u; _Float16 h[2]; } q;
    q.u = *(const unsigned*)(Y + (size_t)kidx * MROWS + rowB + rp);
    yt[(rp    ) * YT_PITCH + kidx] = q.h[0];
    yt[(rp + 1) * YT_PITCH + kidx] = q.h[1];
  }
  __syncthreads();
  int m = lane & 15, hi8 = (lane >> 4) * 8;
  v8f acc[4] = {};
#pragma unroll
  for (int k0 = 0; k0 < 64; k0 += 32) {
    const _Float16* ap = yt + (wv * 16 + m) * YT_PITCH + k0 + hi8;
    union { uint4 u[2]; v16h h; } ua;
    ua.u[0] = *(const uint4*)(ap);
    ua.u[1] = *(const uint4*)(ap + 16);
#pragma unroll
    for (int t = 0; t < 4; ++t) {
      v16h b = load_b_packed(BdP, (k0 >> 5) * 4 + t, lane);
      acc[t] = __builtin_amdgcn_wmma_f32_16x16x32_f16(false, ua.h, false, b,
                                                      (short)0, acc[t], false, false);
    }
  }
  float* dsw = dstage + wv * (16 * 64);
  int mh = (lane >> 4) * 8, n = lane & 15;
#pragma unroll
  for (int t = 0; t < 4; ++t)
#pragma unroll
    for (int r = 0; r < 8; ++r)
      dsw[(mh + r) * 64 + t * 16 + n] = acc[t][r];
  __syncthreads();
#pragma unroll
  for (int j = 0; j < 13; ++j) {           // 13*32 = 416 mags per wave tile
    int idx = j * 32 + lane;
    int mm = idx / KK, k = idx % KK;
    float re = dsw[mm * 64 + k], im = dsw[mm * 64 + KK + k];
    float mag = sqrtf(re * re + im * im);
    int grow  = rowB + wv * 16 + mm;        // = (b*16+c)*4096 + p
    int cch   = (grow >> 12) & 15;
    int pixel = (grow >> 16) * HW + (grow & (HW - 1));
    F[(size_t)pixel * FEAT + cch * KK + k] = (_Float16)mag;
  }
}

// MLP GEMM: 256-thread blocks (8 waves), 128 rows x 64 cols per block.
// Packed B (contiguous 4KB per k-step) staged to LDS with
// global_load_async_to_lds_b128, waited with s_wait_asynccnt. A via 2x b128.
__global__ void gemm_h(const _Float16* __restrict__ A, const _Float16* __restrict__ WP,
                       const float* __restrict__ bias, _Float16* __restrict__ Out,
                       int Kd, int N, int relu) {
  __shared__ __align__(16) _Float16 bt[4 * 32 * 16];   // 4 KB
  int tid = threadIdx.x, lane = tid & 31, wv = tid >> 5;
  int row0 = blockIdx.x * 128 + wv * 16;
  int n0 = blockIdx.y * 64;
  int ntn = N >> 4;
  int m = lane & 15, hi8 = (lane >> 4) * 8;
  const _Float16* arow = A + (size_t)(row0 + m) * Kd;
  v8f acc[4] = {};
  for (int k0 = 0; k0 < Kd; k0 += 32) {
    __syncthreads();
    {
      const _Float16* gsrc = WP + ((size_t)(k0 >> 5) * ntn + (n0 >> 4)) * 512 + tid * 8;
      const _Float16* ldst = bt + tid * 8;
      asm volatile("global_load_async_to_lds_b128 %0, %1, off"
                   :: "v"((unsigned)(size_t)ldst), "v"(gsrc) : "memory");
    }
    union { uint4 u[2]; v16h h; } ua;
    const _Float16* ap = arow + k0 + hi8;
    ua.u[0] = *(const uint4*)ap;
    ua.u[1] = *(const uint4*)(ap + 16);
    asm volatile("s_wait_asynccnt 0x0" ::: "memory");
    __syncthreads();
#pragma unroll
    for (int t = 0; t < 4; ++t) {
      union { uint4 u[2]; v16h h; } ub;
      const _Float16* bp = bt + (t * 32 + lane) * 16;
      ub.u[0] = *(const uint4*)bp;
      ub.u[1] = *(const uint4*)(bp + 8);
      acc[t] = __builtin_amdgcn_wmma_f32_16x16x32_f16(false, ua.h, false, ub.h,
                                                      (short)0, acc[t], false, false);
    }
  }
  int mh = (lane >> 4) * 8, n = lane & 15;
#pragma unroll
  for (int t = 0; t < 4; ++t) {
    int col = n0 + t * 16 + n;
    float bv = bias[col];
#pragma unroll
    for (int r = 0; r < 8; ++r) {
      float v = acc[t][r] + bv;
      if (relu) v = fmaxf(v, 0.f);
      Out[(size_t)(row0 + mh + r) * N + col] = (_Float16)v;
    }
  }
}

// Final layer: K=256, N padded 7->16, writes f32 logits transposed (B,NC,H,W).
__global__ void gemm_final(const _Float16* __restrict__ A, const _Float16* __restrict__ WP,
                           const float* __restrict__ bias, float* __restrict__ logits) {
  int lane = threadIdx.x;
  int row0 = blockIdx.x * 16;
  int m = lane & 15, hi8 = (lane >> 4) * 8;
  const _Float16* arow = A + (size_t)(row0 + m) * HID;
  v8f acc = {};
  for (int k0 = 0; k0 < HID; k0 += 32) {
    v16h a = load_a_b128(arow, k0, hi8);
    v16h b = load_b_packed(WP, k0 >> 5, lane);
    acc = __builtin_amdgcn_wmma_f32_16x16x32_f16(false, a, false, b,
                                                 (short)0, acc, false, false);
  }
  int n = lane & 15, mh = (lane >> 4) * 8;
  if (n < NCLS) {
    float bv = bias[n];
#pragma unroll
    for (int r = 0; r < 8; ++r) {
      int pixel = row0 + mh + r;
      int b = pixel >> 12, p = pixel & (HW - 1);
      logits[((size_t)b * NCLS + n) * HW + p] = acc[r] + bv;
    }
  }
}

// f32 -> f16 weight repack into fragment-native packed-B layout:
// out[((kt*(N/16)+nt)*32+lane)*16+i] = W[kt*32+(lane>>4)*8+i+(i>=8?8:0)][nt*16+(lane&15)]
// (srcN = real column count; columns >= srcN zero-padded).
__global__ void pack_w(const float* __restrict__ Wsrc, _Float16* __restrict__ out,
                       int Kd, int N, int srcN) {
  int idx = blockIdx.x * 256 + threadIdx.x;
  if (idx >= Kd * N) return;
  int ii = idx & 15;
  int lane = (idx >> 4) & 31;
  int tile = idx >> 9;
  int ntn = N >> 4;
  int nt = tile % ntn, kt = tile / ntn;
  int k = kt * 32 + (lane >> 4) * 8 + ii + (ii >= 8 ? 8 : 0);
  int n = nt * 16 + (lane & 15);
  float v = (n < srcN) ? Wsrc[(size_t)k * srcN + n] : 0.f;
  out[idx] = (_Float16)v;
}

// ------------------------------ launcher -----------------------------------
extern "C" void kernel_launch(void* const* d_in, const int* in_sizes, int n_in,
                              void* d_out, int out_size, void* d_ws, size_t ws_size,
                              hipStream_t stream) {
  (void)in_sizes; (void)n_in; (void)out_size; (void)ws_size;
  const float* x = (const float*)d_in[0];
  const float* om_w[3] = {(const float*)d_in[1], (const float*)d_in[3], (const float*)d_in[5]};
  const float* om_b[3] = {(const float*)d_in[2], (const float*)d_in[4], (const float*)d_in[6]};
  const float* al_w[3] = {(const float*)d_in[7], (const float*)d_in[9], (const float*)d_in[11]};
  const float* al_b[3] = {(const float*)d_in[8], (const float*)d_in[10], (const float*)d_in[12]};
  const float* hy_w[4] = {(const float*)d_in[13], (const float*)d_in[15],
                          (const float*)d_in[17], (const float*)d_in[19]};
  const float* hy_b[4] = {(const float*)d_in[14], (const float*)d_in[16],
                          (const float*)d_in[18], (const float*)d_in[20]};
  const float* wy = (const float*)d_in[21];
  const float* ro_w[4] = {(const float*)d_in[22], (const float*)d_in[24],
                          (const float*)d_in[26], (const float*)d_in[28]};
  const float* ro_b[4] = {(const float*)d_in[23], (const float*)d_in[25],
                          (const float*)d_in[27], (const float*)d_in[29]};

  char* w = (char*)d_ws;
  size_t off = 0;
  auto carve = [&](size_t bytes) { size_t o = off; off += (bytes + 255) & ~(size_t)255; return o; };
  const size_t CONV_B = (size_t)BB * CC * HW * 4;
  float*    bufA = (float*)(w + carve(CONV_B));
  float*    bufB = (float*)(w + carve(CONV_B));
  float*    omB  = (float*)(w + carve(CONV_B));
  float*    alB  = (float*)(w + carve(CONV_B));
  float*    hy0B = (float*)(w + carve(CONV_B));
  _Float16* wdft = (_Float16*)(w + carve(4096 * 2));
  _Float16* w1h  = (_Float16*)(w + carve((size_t)FEAT * HID * 2));
  _Float16* w2h  = (_Float16*)(w + carve((size_t)HID * HID * 2));
  _Float16* w3h  = (_Float16*)(w + carve((size_t)HID * HID * 2));
  _Float16* w4h  = (_Float16*)(w + carve((size_t)HID * 16 * 2));
  _Float16* yb   = (_Float16*)(w + carve((size_t)64 * MROWS * 2));   // 128 MB
  _Float16* Fb   = (_Float16*)(w + carve((size_t)MPIX * FEAT * 2));  // 52 MB
  _Float16* a1   = (_Float16*)(w + carve((size_t)MPIX * HID * 2));
  _Float16* a2   = (_Float16*)(w + carve((size_t)MPIX * HID * 2));

  dim3 cgrid(SS / 4, CC, BB);
  conv3x3<<<cgrid, 256, 0, stream>>>(x,    om_w[0], om_b[0], bufA, 1,  0);
  conv3x3<<<cgrid, 256, 0, stream>>>(bufA, om_w[1], om_b[1], bufB, CC, 0);
  conv3x3<<<cgrid, 256, 0, stream>>>(bufB, om_w[2], om_b[2], omB,  CC, 0);
  conv3x3<<<cgrid, 256, 0, stream>>>(x,    al_w[0], al_b[0], bufA, 1,  0);
  conv3x3<<<cgrid, 256, 0, stream>>>(bufA, al_w[1], al_b[1], bufB, CC, 0);
  conv3x3<<<cgrid, 256, 0, stream>>>(bufB, al_w[2], al_b[2], alB,  CC, 0);
  conv3x3<<<cgrid, 256, 0, stream>>>(x,    hy_w[0], hy_b[0], bufA, 1,  0);
  conv3x3<<<cgrid, 256, 0, stream>>>(bufA, hy_w[1], hy_b[1], bufB, CC, 0);
  conv3x3<<<cgrid, 256, 0, stream>>>(bufB, hy_w[2], hy_b[2], bufA, CC, 0);
  conv3x3<<<cgrid, 256, 0, stream>>>(bufA, hy_w[3], hy_b[3], hy0B, CC, 1);

  dft_init<<<1, 256, 0, stream>>>(wdft);
  pack_w<<<(FEAT * HID + 255) / 256, 256, 0, stream>>>(ro_w[0], w1h, FEAT, HID, HID);
  pack_w<<<(HID * HID + 255) / 256, 256, 0, stream>>>(ro_w[1], w2h, HID, HID, HID);
  pack_w<<<(HID * HID + 255) / 256, 256, 0, stream>>>(ro_w[2], w3h, HID, HID, HID);
  pack_w<<<(HID * 16 + 255) / 256, 256, 0, stream>>>(ro_w[3], w4h, HID, 16, NCLS);

  ode_scan<<<BB * CC, 256, 0, stream>>>(hy0B, omB, alB, wy, yb);

  dft_mag<<<MROWS / 128, 256, 0, stream>>>(yb, wdft, Fb);

  gemm_h<<<dim3(MPIX / 128, HID / 64), 256, 0, stream>>>(Fb, w1h, ro_b[0], a1, FEAT, HID, 1);
  gemm_h<<<dim3(MPIX / 128, HID / 64), 256, 0, stream>>>(a1, w2h, ro_b[1], a2, HID, HID, 1);
  gemm_h<<<dim3(MPIX / 128, HID / 64), 256, 0, stream>>>(a2, w3h, ro_b[2], a1, HID, HID, 1);
  gemm_final<<<MPIX / 16, 32, 0, stream>>>(a1, w4h, ro_b[3], (float*)d_out);
}
